// HF_T5_Partial_34497177321886
// MI455X (gfx1250) — matrix operand order repaired and checked by reading, not compiled, not run on hardware
//
#include <hip/hip_runtime.h>

// ---------------------------------------------------------------------------
// CDNA5 (gfx1250) T5 encoder block forward.  wave32, WMMA f32_16x16x32_f16.
// ---------------------------------------------------------------------------

typedef _Float16 half_t;
typedef __attribute__((ext_vector_type(16))) _Float16 v16h;
typedef __attribute__((ext_vector_type(8)))  _Float16 v8h;
typedef __attribute__((ext_vector_type(8)))  float    v8f;
typedef __attribute__((ext_vector_type(4)))  float    v4f;

#define WMMA_F16(a, b, c) \
  __builtin_amdgcn_wmma_f32_16x16x32_f16(false, (a), false, (b), (short)0, (c), false, false)

static __device__ __forceinline__ v8f zero8() {
  v8f z;
#pragma unroll
  for (int i = 0; i < 8; ++i) z[i] = 0.0f;
  return z;
}

// A-fragment: 16x32 f16, M rows of A (row-major, leading dim lda, K-contig).
// lanes 0-15: halves 0-15 = K kb+0..15
// lanes16-31: halves 0-15 = K kb+16..31
static __device__ __forceinline__ v16h load_a(const half_t* __restrict__ A,
                                              int lda, int row0, int kb) {
  const int lane = threadIdx.x & 31;
  const int hi = lane >> 4;
  const half_t* p = A + (size_t)(row0 + (lane & 15)) * lda + kb + hi * 16;
  v8h lo = *(const v8h*)p;
  v8h hp = *(const v8h*)(p + 8);
  v16h out;
#pragma unroll
  for (int i = 0; i < 8; ++i) { out[i] = lo[i]; out[i + 8] = hp[i]; }
  return out;
}

// B-fragment for C = X @ W^T: B[k][n] = W[n][k].  Lane n holds W row n,
// 16 contiguous K halves (lanes 0-15: K kb+0..15, lanes 16-31: K kb+16..31).
static __device__ __forceinline__ v16h load_b(const half_t* __restrict__ W,
                                              int ldw, int n0, int kb) {
  const int lane = threadIdx.x & 31;
  const int hi = lane >> 4;
  return *(const v16h*)(W + (size_t)(n0 + (lane & 15)) * ldw + kb + hi * 16);
}

// ---------------------------------------------------------------------------
// f32 -> f16 weight conversion
// ---------------------------------------------------------------------------
__global__ void k_f32_to_f16(const float* __restrict__ s, half_t* __restrict__ d, int n) {
  int i = blockIdx.x * 256 + threadIdx.x;
  if (i < n) d[i] = (half_t)s[i];
}

// ---------------------------------------------------------------------------
// RMSNorm: one row (D=512) per 256-thread block, f16 output.
// ---------------------------------------------------------------------------
__global__ __launch_bounds__(256) void k_rmsnorm(const float* __restrict__ X,
                                                 const float* __restrict__ w,
                                                 half_t* __restrict__ H, int D) {
  const int row = blockIdx.x;
  const float* x = X + (size_t)row * D;
  float ss = 0.0f;
  for (int i = threadIdx.x; i < D; i += 256) { float v = x[i]; ss += v * v; }
#pragma unroll
  for (int m = 1; m < 32; m <<= 1) ss += __shfl_xor(ss, m, 32);
  __shared__ float red[8];
  if ((threadIdx.x & 31) == 0) red[threadIdx.x >> 5] = ss;
  __syncthreads();
  float tot = 0.0f;
#pragma unroll
  for (int i = 0; i < 8; ++i) tot += red[i];
  const float rs = rsqrtf(tot / (float)D + 1e-6f);
  for (int i = threadIdx.x; i < D; i += 256)
    H[(size_t)row * D + i] = (half_t)(x[i] * rs * w[i]);
}

// ---------------------------------------------------------------------------
// GEMM  C(f16)[M,N] = A(f16)[M,K] @ W(f16)[N,K]^T   (+ optional ReLU)
// block = 8 waves, each wave: 16 rows x 64 cols.
// ---------------------------------------------------------------------------
__global__ __launch_bounds__(256) void k_gemm_f16(const half_t* __restrict__ A,
                                                  const half_t* __restrict__ W,
                                                  half_t* __restrict__ C,
                                                  int M, int N, int K, int relu) {
  const int widx = threadIdx.x >> 5;
  const int lane = threadIdx.x & 31;
  const int row0 = (blockIdx.x * 8 + widx) * 16;
  const int n0 = blockIdx.y * 64;
  v8f acc[4];
#pragma unroll
  for (int t = 0; t < 4; ++t) acc[t] = zero8();

  for (int kb = 0; kb < K; kb += 32) {
    v16h a = load_a(A, K, row0, kb);
#pragma unroll
    for (int t = 0; t < 4; ++t) {
      v16h b = load_b(W, K, n0 + t * 16, kb);
      acc[t] = WMMA_F16(a, b, acc[t]);
    }
  }
  const int hi = lane >> 4, col = lane & 15;
#pragma unroll
  for (int t = 0; t < 4; ++t)
#pragma unroll
    for (int j = 0; j < 8; ++j) {
      float v = acc[t][j];
      if (relu) v = v > 0.0f ? v : 0.0f;
      C[(size_t)(row0 + j + 8 * hi) * N + n0 + t * 16 + col] = (half_t)v;
    }
}

// ---------------------------------------------------------------------------
// GEMM writing V transposed: VT[(b*8+h)*64 + d][s]  (16B vector stores).
// M=8192, N=512, K=512, shapes hardcoded for the attention layout.
// ---------------------------------------------------------------------------
__global__ __launch_bounds__(256) void k_gemm_vt(const half_t* __restrict__ A,
                                                 const half_t* __restrict__ W,
                                                 half_t* __restrict__ VT,
                                                 int M, int N, int K) {
  const int widx = threadIdx.x >> 5;
  const int lane = threadIdx.x & 31;
  const int row0 = (blockIdx.x * 8 + widx) * 16;
  const int n0 = blockIdx.y * 64;
  v8f acc[4];
#pragma unroll
  for (int t = 0; t < 4; ++t) acc[t] = zero8();

  for (int kb = 0; kb < K; kb += 32) {
    v16h a = load_a(A, K, row0, kb);
#pragma unroll
    for (int t = 0; t < 4; ++t) {
      v16h b = load_b(W, K, n0 + t * 16, kb);
      acc[t] = WMMA_F16(a, b, acc[t]);
    }
  }
  const int hi = lane >> 4, col = lane & 15;
  const int m = row0 + 8 * hi;            // j spans m..m+7 (contiguous s)
  const int bq = m >> 10;                 // batch
  const int s = m & 1023;
#pragma unroll
  for (int t = 0; t < 4; ++t) {
    const int n = n0 + t * 16 + col;      // n = h*64 + d
    v8h o;
#pragma unroll
    for (int j = 0; j < 8; ++j) o[j] = (half_t)acc[t][j];
    half_t* dst = VT + (((size_t)(bq * 8 + (n >> 6)) * 64 + (n & 63)) * 1024 + s);
    *(v8h*)dst = o;
  }
}

// ---------------------------------------------------------------------------
// GEMM  C(f32)[M,N] = R(f32)[M,N] + A(f16)[M,K] @ W(f16)[N,K]^T  (residual)
// ---------------------------------------------------------------------------
__global__ __launch_bounds__(256) void k_gemm_res(const half_t* __restrict__ A,
                                                  const half_t* __restrict__ W,
                                                  const float* __restrict__ R,
                                                  float* __restrict__ C,
                                                  int M, int N, int K) {
  const int widx = threadIdx.x >> 5;
  const int lane = threadIdx.x & 31;
  const int row0 = (blockIdx.x * 8 + widx) * 16;
  const int n0 = blockIdx.y * 64;
  v8f acc[4];
#pragma unroll
  for (int t = 0; t < 4; ++t) acc[t] = zero8();

  for (int kb = 0; kb < K; kb += 32) {
    v16h a = load_a(A, K, row0, kb);
#pragma unroll
    for (int t = 0; t < 4; ++t) {
      v16h b = load_b(W, K, n0 + t * 16, kb);
      acc[t] = WMMA_F16(a, b, acc[t]);
    }
  }
  const int hi = lane >> 4, col = lane & 15;
#pragma unroll
  for (int t = 0; t < 4; ++t)
#pragma unroll
    for (int j = 0; j < 8; ++j) {
      const size_t idx = (size_t)(row0 + j + 8 * hi) * N + n0 + t * 16 + col;
      C[idx] = R[idx] + acc[t][j];
    }
}

// ---------------------------------------------------------------------------
// Flash attention.  Per wave: one (b,h), 16 query rows, online softmax over
// 32-key blocks.  Computes S^T = K @ Q^T so the exp'd score fragment IS the
// A-fragment for p @ V (no LDS, no transpose).  Bias added from global f32.
// ---------------------------------------------------------------------------
__global__ __launch_bounds__(256) void k_attn(const half_t* __restrict__ Q,
                                              const half_t* __restrict__ Kx,
                                              const half_t* __restrict__ VT,
                                              const float* __restrict__ bias,
                                              half_t* __restrict__ CTX) {
  const int D = 512, S = 1024, HD = 64;
  const int widx = threadIdx.x >> 5;
  const int lane = threadIdx.x & 31;
  const int wg = blockIdx.x * 8 + widx;
  const int bh = wg >> 6;                  // (b*8 + h)
  const int q0 = (wg & 63) * 16;           // query tile base
  const int b = bh >> 3, h = bh & 7;
  const int col = lane & 15, hi = lane >> 4;

  const half_t* Qp = Q + (size_t)b * S * D + h * HD;
  const half_t* Kp = Kx + (size_t)b * S * D + h * HD;
  const half_t* Vp = VT + (size_t)bh * HD * S;               // [HD][S]
  const float* Bp = bias + ((size_t)bh * S + q0) * S;        // [16 q][S k]

  // B-fragments of Q^T (lane = query column), loaded once.
  v16h qb[2];
#pragma unroll
  for (int c = 0; c < 2; ++c)
    qb[c] = *(const v16h*)(Qp + (size_t)(q0 + col) * D + c * 32 + hi * 16);

  v8f acc[4];
#pragma unroll
  for (int t = 0; t < 4; ++t) acc[t] = zero8();
  float mrow = -1e30f, lrow = 0.0f;        // stats for query q0+col

  for (int k0 = 0; k0 < S; k0 += 32) {
    // ---- S^T tiles: two 16(key)x16(query) fragments --------------------
    v8f sc0 = zero8(), sc1 = zero8();
#pragma unroll
    for (int c = 0; c < 2; ++c) {
      v16h a0 = load_a(Kp, D, k0, c * 32);
      v16h a1 = load_a(Kp, D, k0 + 16, c * 32);
      sc0 = WMMA_F16(a0, qb[c], sc0);
      sc1 = WMMA_F16(a1, qb[c], sc1);
    }
    // ---- bias: lane reads 2x 32B contiguous runs of its query row ------
    const float* bp = Bp + (size_t)col * S + k0 + hi * 8;
    if (k0 + 32 < S) __builtin_prefetch(bp + 32, 0, 1);
    v4f ba0 = *(const v4f*)bp;
    v4f ba1 = *(const v4f*)(bp + 4);
    v4f bb0 = *(const v4f*)(bp + 16);
    v4f bb1 = *(const v4f*)(bp + 20);
    float s0[8], s1[8];
#pragma unroll
    for (int j = 0; j < 4; ++j) {
      s0[j]     = sc0[j]     + ba0[j];
      s0[j + 4] = sc0[j + 4] + ba1[j];
      s1[j]     = sc1[j]     + bb0[j];
      s1[j + 4] = sc1[j + 4] + bb1[j];
    }
    // ---- online softmax (per-lane over 16 keys, pair-combine via xor16) -
    float mx = s0[0];
#pragma unroll
    for (int j = 1; j < 8; ++j) mx = fmaxf(mx, s0[j]);
#pragma unroll
    for (int j = 0; j < 8; ++j) mx = fmaxf(mx, s1[j]);
    mx = fmaxf(mx, __shfl_xor(mx, 16, 32));
    const float mnew = fmaxf(mrow, mx);
    const float scale = __expf(mrow - mnew);
    mrow = mnew;

    v16h pa;                               // == A-fragment of p, for free
    float ls = 0.0f;
#pragma unroll
    for (int j = 0; j < 8; ++j) {
      float p = __expf(s0[j] - mnew); ls += p; pa[j] = (half_t)p;
    }
#pragma unroll
    for (int j = 0; j < 8; ++j) {
      float p = __expf(s1[j] - mnew); ls += p; pa[8 + j] = (half_t)p;
    }
    ls += __shfl_xor(ls, 16, 32);
    lrow = lrow * scale + ls;

    // rescale accumulator rows (row q = j + 8*hi -> stat lives in lane j+8*hi)
    float srow[8];
#pragma unroll
    for (int j = 0; j < 8; ++j) srow[j] = __shfl(scale, j + 8 * hi, 32);
#pragma unroll
    for (int t = 0; t < 4; ++t)
#pragma unroll
      for (int j = 0; j < 8; ++j) acc[t][j] *= srow[j];

    // ---- ctx += p @ V  (B-frags contiguous from transposed V) ----------
#pragma unroll
    for (int t = 0; t < 4; ++t) {
      const half_t* vp = Vp + (size_t)(t * 16 + col) * S + k0 + hi * 8;
      v8h v0 = *(const v8h*)vp;
      v8h v1 = *(const v8h*)(vp + 16);
      v16h vb;
#pragma unroll
      for (int i = 0; i < 8; ++i) { vb[i] = v0[i]; vb[i + 8] = v1[i]; }
      acc[t] = WMMA_F16(pa, vb, acc[t]);
    }
  }

  // ---- normalize and store ctx as f16 [B,S,H*HD] ------------------------
  float lr[8];
#pragma unroll
  for (int j = 0; j < 8; ++j) lr[j] = __shfl(lrow, j + 8 * hi, 32);
#pragma unroll
  for (int t = 0; t < 4; ++t)
#pragma unroll
    for (int j = 0; j < 8; ++j) {
      const int q = q0 + j + 8 * hi;
      CTX[((size_t)b * S + q) * D + h * HD + t * 16 + col] =
          (half_t)(acc[t][j] / lr[j]);
    }
}

// ---------------------------------------------------------------------------
// Host launcher
// ---------------------------------------------------------------------------
extern "C" void kernel_launch(void* const* d_in, const int* in_sizes, int n_in,
                              void* d_out, int out_size, void* d_ws, size_t ws_size,
                              hipStream_t stream) {
  (void)in_sizes; (void)n_in; (void)out_size; (void)ws_size;
  const int B = 8, S = 1024, D = 512, F = 2048;
  const int M = B * S;  // 8192

  const float* pWk  = (const float*)d_in[0];  // primals_1 [D,D]
  const float* pWo  = (const float*)d_in[1];  // primals_2 [D,D]
  const float* pWq  = (const float*)d_in[2];  // primals_3 [D,D]
  const float* pWv  = (const float*)d_in[3];  // primals_4 [D,D]
  const float* pG1  = (const float*)d_in[4];  // primals_5 [D]
  const float* pW6  = (const float*)d_in[5];  // primals_6 [F,D]
  const float* pW7  = (const float*)d_in[6];  // primals_7 [D,F]
  const float* pG2  = (const float*)d_in[7];  // primals_8 [D]
  const float* pX   = (const float*)d_in[8];  // primals_9 [B,S,D]
  const float* pBia = (const float*)d_in[9];  // primals_10 [B,H,S,S]

  // workspace carve-up (1KB aligned)
  char* ws = (char*)d_ws;
  size_t off = 0;
  auto carve = [&](size_t bytes) -> char* {
    char* p = ws + off;
    off += (bytes + 1023) & ~(size_t)1023;
    return p;
  };
  half_t* Wq  = (half_t*)carve((size_t)D * D * 2);
  half_t* Wk  = (half_t*)carve((size_t)D * D * 2);
  half_t* Wv  = (half_t*)carve((size_t)D * D * 2);
  half_t* Wo  = (half_t*)carve((size_t)D * D * 2);
  half_t* W6  = (half_t*)carve((size_t)F * D * 2);
  half_t* W7  = (half_t*)carve((size_t)D * F * 2);
  half_t* h1  = (half_t*)carve((size_t)M * D * 2);
  half_t* q   = (half_t*)carve((size_t)M * D * 2);
  half_t* kx  = (half_t*)carve((size_t)M * D * 2);
  half_t* vt  = (half_t*)carve((size_t)M * D * 2);  // [64][64][1024]
  half_t* ctx = (half_t*)carve((size_t)M * D * 2);
  float*  x1  = (float*) carve((size_t)M * D * 4);
  half_t* h2  = (half_t*)carve((size_t)M * D * 2);
  half_t* ff  = (half_t*)carve((size_t)M * F * 2);

  // weight conversion
  const int nDD = D * D, nFD = F * D;
  k_f32_to_f16<<<(nDD + 255) / 256, 256, 0, stream>>>(pWq, Wq, nDD);
  k_f32_to_f16<<<(nDD + 255) / 256, 256, 0, stream>>>(pWk, Wk, nDD);
  k_f32_to_f16<<<(nDD + 255) / 256, 256, 0, stream>>>(pWv, Wv, nDD);
  k_f32_to_f16<<<(nDD + 255) / 256, 256, 0, stream>>>(pWo, Wo, nDD);
  k_f32_to_f16<<<(nFD + 255) / 256, 256, 0, stream>>>(pW6, W6, nFD);
  k_f32_to_f16<<<(nFD + 255) / 256, 256, 0, stream>>>(pW7, W7, nFD);

  // attention sublayer
  k_rmsnorm<<<M, 256, 0, stream>>>(pX, pG1, h1, D);
  dim3 gDD(M / 128, D / 64);
  k_gemm_f16<<<gDD, 256, 0, stream>>>(h1, Wq, q,  M, D, D, 0);
  k_gemm_f16<<<gDD, 256, 0, stream>>>(h1, Wk, kx, M, D, D, 0);
  k_gemm_vt <<<gDD, 256, 0, stream>>>(h1, Wv, vt, M, D, D);
  k_attn<<<512, 256, 0, stream>>>(q, kx, vt, pBia, ctx);
  k_gemm_res<<<gDD, 256, 0, stream>>>(ctx, Wo, pX, x1, M, D, D);

  // FFN sublayer
  k_rmsnorm<<<M, 256, 0, stream>>>(x1, pG2, h2, D);
  dim3 gDF(M / 128, F / 64);
  k_gemm_f16<<<gDF, 256, 0, stream>>>(h2, W6, ff, M, F, D, 1);
  k_gemm_res<<<gDD, 256, 0, stream>>>(ff, W7, x1, (float*)d_out, M, D, F);
}
